// SAKEEnergyModel_37580963840558
// MI455X (gfx1250) — matrix operand order repaired and checked by reading, not compiled, not run on hardware
//
#include <hip/hip_runtime.h>
#include <hip/hip_bf16.h>

// ---------------------------------------------------------------------------
// SAKE GNN energy model for gfx1250 (MI455X).
// All K>=64 GEMMs run on v_wmma_f32_16x16x32_f16 (f16 operands, f32 accum).
// Segment reductions use global float atomics. x_mixing GEMM fuses the
// tanh + direction scatter so coeff (E x 256) is never materialized.
// ---------------------------------------------------------------------------

#define NNODES 10000
#define NEDGES 160000
#define NGRAPH 100
#define FIN    16
#define HIDD   64
#define HEADS  4
#define NCOEF  256
#define RBFN   50
#define DEPTH  6

typedef _Float16 half_t;
typedef __attribute__((ext_vector_type(16))) _Float16 v16h;
typedef __attribute__((ext_vector_type(8)))  float    v8f;

struct __align__(16) U4 { unsigned int x, y, z, w; };
struct __align__(16) F4 { float x, y, z, w; };
struct __align__(8)  H4 { half_t a, b, c, d; };

union Frag { v16h v; U4 q[2]; };

__device__ __forceinline__ float siluf(float x) { return x / (1.f + __expf(-x)); }
__device__ __forceinline__ float fixfin(float m) {
  return (m >= -3.0e38f && m <= 3.0e38f) ? m : 0.f;
}
__device__ __forceinline__ void atomicMaxF(float* a, float v) {
  if (v >= 0.f) atomicMax((int*)a, __float_as_int(v));
  else          atomicMin((unsigned int*)a, __float_as_uint(v));
}

// ---------------------------------------------------------------------------
// Utility kernels
// ---------------------------------------------------------------------------
__global__ void fill_f32(float* p, float v, int n) {
  int i = blockIdx.x * blockDim.x + threadIdx.x;
  if (i < n) p[i] = v;
}

// Pack fp32 weight (K x N row-major) into WMMA-B f16 operand layout, padded to
// KP x NP.  For tile (nb,kb): lane L (n=L%16, g=L/16), vgpr v holds
// K = kb*32 + g*16 + 2v + h.  Flat: halves[(((nb*(KP/32)+kb)*32)+L)*16 + 2v+h]
__global__ void pack_wmma_b(const float* __restrict__ W, int K, int N,
                            int KP, int NP, half_t* __restrict__ out) {
  int t = blockIdx.x * blockDim.x + threadIdx.x;   // one half2 each
  int total = (KP * NP) / 2;
  if (t >= total) return;
  int v  = t & 7;
  int L  = (t >> 3) & 31;
  int kb = (t >> 8) % (KP / 32);
  int nb = t / (256 * (KP / 32));
  int n  = nb * 16 + (L & 15);
  int k0 = kb * 32 + (L >> 4) * 16 + 2 * v;
  float w0 = (k0     < K && n < N) ? W[(size_t)k0 * N + n]       : 0.f;
  float w1 = (k0 + 1 < K && n < N) ? W[(size_t)(k0 + 1) * N + n] : 0.f;
  out[2 * t]     = (half_t)w0;
  out[2 * t + 1] = (half_t)w1;
}

__global__ void pack_bias_pad(const float* b, int N, int NP, float* out) {
  int i = blockIdx.x * blockDim.x + threadIdx.x;
  if (i < NP) out[i] = (i < N) ? b[i] : 0.f;
}

// ---------------------------------------------------------------------------
// Generic row-panel WMMA GEMM: out[rows x NP] = epi(A[rows x KP] @ B + bias)
// A is f16 row-major (zero padded to KP).  32 rows / block, 8 waves,
// each wave owns 16x16 tiles.  epi: 0=none 1=silu 2=tanh.  resid added last.
// ---------------------------------------------------------------------------
template <int KP, int NP>
__global__ __launch_bounds__(256) void gemm_rows(
    const half_t* __restrict__ A, const half_t* __restrict__ Bw,
    const float* __restrict__ bias, int rows,
    float* __restrict__ outf, half_t* __restrict__ out16,
    const float* __restrict__ resid, int epi) {
  __shared__ U4 AsU[32 * KP / 8];
  half_t* As = (half_t*)AsU;
  const int row0 = blockIdx.x * 32;
  const int t = threadIdx.x;
  for (int i = t; i < 32 * KP / 8; i += 256) {
    const int r = i / (KP / 8);
    U4 val;
    if (row0 + r < rows) {
      val = *(const U4*)(A + (size_t)(row0 + r) * KP + (i % (KP / 8)) * 8);
    } else {
      val.x = val.y = val.z = val.w = 0u;
    }
    AsU[i] = val;
  }
  __syncthreads();

  const int w  = t >> 5;
  const int L  = t & 31;
  const int hf = L >> 4;
  const int ln = L & 15;
  const int NT = 2 * (NP / 16);
  for (int tile = w; tile < NT; tile += 8) {
    const int rh = tile & 1;
    const int ct = tile >> 1;
    v8f acc = {};
    const int rloc = rh * 16 + ln;
    for (int kb = 0; kb < KP / 32; ++kb) {
      Frag a, b;
      const half_t* ap = As + rloc * KP + kb * 32 + hf * 16;
      a.q[0] = *(const U4*)(ap);
      a.q[1] = *(const U4*)(ap + 8);
      const half_t* bp = Bw + (((size_t)ct * (KP / 32) + kb) * 32 + L) * 16;
      b.q[0] = ((const U4*)bp)[0];
      b.q[1] = ((const U4*)bp)[1];
      acc = __builtin_amdgcn_wmma_f32_16x16x32_f16(
          false, a.v, false, b.v, (short)0, acc, false, false);
    }
    const int col = ct * 16 + ln;
    const float bc = bias ? bias[col] : 0.f;
#pragma unroll
    for (int v = 0; v < 8; ++v) {
      const int r = row0 + rh * 16 + v + 8 * hf;
      if (r >= rows) continue;
      float x = acc[v] + bc;
      if (epi == 1) x = siluf(x);
      else if (epi == 2) x = tanhf(x);
      if (resid) x += resid[(size_t)r * NP + col];
      if (outf)  outf[(size_t)r * NP + col]  = x;
      if (out16) out16[(size_t)r * NP + col] = (half_t)x;
    }
  }
}

// ---------------------------------------------------------------------------
// x_mixing WMMA GEMM fused with tanh + directional scatter:
//   coeff = tanh(he_att @ Wx);  comb_vec[recv[e], c, a] += coeff[e,c]*dir[e,a]
// ---------------------------------------------------------------------------
__global__ __launch_bounds__(256) void xmix_scatter(
    const half_t* __restrict__ A, const half_t* __restrict__ Bw,
    const int* __restrict__ rcv, const float* __restrict__ edge4,
    float* __restrict__ comb_vec, int rows) {
  constexpr int KP = 256;
  __shared__ U4 AsU[32 * KP / 8];
  __shared__ float dirS[32][4];
  __shared__ int rS[32];
  half_t* As = (half_t*)AsU;
  const int row0 = blockIdx.x * 32;
  const int t = threadIdx.x;
  if (t < 32)  rS[t] = (row0 + t < rows) ? rcv[row0 + t] : 0;
  if (t < 128) {
    int e = t >> 2, a = t & 3;
    dirS[e][a] = (a < 3 && row0 + e < rows) ? edge4[(size_t)(row0 + e) * 4 + a] : 0.f;
  }
  for (int i = t; i < 32 * KP / 8; i += 256) {
    const int r = i / (KP / 8);
    U4 val;
    if (row0 + r < rows) {
      val = *(const U4*)(A + (size_t)(row0 + r) * KP + (i % (KP / 8)) * 8);
    } else {
      val.x = val.y = val.z = val.w = 0u;
    }
    AsU[i] = val;
  }
  __syncthreads();

  const int w  = t >> 5;
  const int L  = t & 31;
  const int hf = L >> 4;
  const int ln = L & 15;
  for (int tile = w; tile < 32; tile += 8) {
    const int rh = tile & 1;
    const int ct = tile >> 1;
    v8f acc = {};
    const int rloc = rh * 16 + ln;
    for (int kb = 0; kb < KP / 32; ++kb) {
      Frag a, b;
      const half_t* ap = As + rloc * KP + kb * 32 + hf * 16;
      a.q[0] = *(const U4*)(ap);
      a.q[1] = *(const U4*)(ap + 8);
      const half_t* bp = Bw + (((size_t)ct * (KP / 32) + kb) * 32 + L) * 16;
      b.q[0] = ((const U4*)bp)[0];
      b.q[1] = ((const U4*)bp)[1];
      acc = __builtin_amdgcn_wmma_f32_16x16x32_f16(
          false, a.v, false, b.v, (short)0, acc, false, false);
    }
    const int col = ct * 16 + ln;
#pragma unroll
    for (int v = 0; v < 8; ++v) {
      const int eloc = rh * 16 + v + 8 * hf;
      if (row0 + eloc >= rows) continue;
      const float val = tanhf(acc[v]);
      const int r = rS[eloc];
      float* dst = comb_vec + ((size_t)r * NCOEF + col) * 3;
      atomicAdd(dst + 0, val * dirS[eloc][0]);
      atomicAdd(dst + 1, val * dirS[eloc][1]);
      atomicAdd(dst + 2, val * dirS[eloc][2]);
    }
  }
}

// ---------------------------------------------------------------------------
// Scalar / pointwise kernels
// ---------------------------------------------------------------------------
__global__ void embed_in_k(const float* __restrict__ h, const float* __restrict__ W,
                           const float* __restrict__ b, float* __restrict__ hn) {
  int t = blockIdx.x * blockDim.x + threadIdx.x;
  if (t >= NNODES * HIDD) return;
  int n = t >> 6, c = t & 63;
  float s = b[c];
  for (int j = 0; j < FIN; ++j) s += h[n * FIN + j] * W[j * HIDD + c];
  hn[t] = s;
}

// h_cat = [hn[s], hn[r]] as f16; edge4 = [dirx,diry,dirz,d]
__global__ void edge_build_k(const float* __restrict__ hn, const float* __restrict__ x,
                             const int* __restrict__ snd, const int* __restrict__ rcv,
                             half_t* __restrict__ EA1, float* __restrict__ edge4) {
  int t = blockIdx.x * blockDim.x + threadIdx.x;
  int e = t >> 4, c = t & 15;
  if (e >= NEDGES) return;
  int s = snd[e], r = rcv[e];
  F4 hs = *(const F4*)(hn + (size_t)s * HIDD + c * 4);
  F4 hr = *(const F4*)(hn + (size_t)r * HIDD + c * 4);
  half_t* o = EA1 + (size_t)e * 128;
  H4 a; a.a = (half_t)hs.x; a.b = (half_t)hs.y; a.c = (half_t)hs.z; a.d = (half_t)hs.w;
  *(H4*)(o + c * 4) = a;
  H4 bb; bb.a = (half_t)hr.x; bb.b = (half_t)hr.y; bb.c = (half_t)hr.z; bb.d = (half_t)hr.w;
  *(H4*)(o + 64 + c * 4) = bb;
  if (c == 0) {
    float dx = x[s * 3 + 0] - x[r * 3 + 0];
    float dy = x[s * 3 + 1] - x[r * 3 + 1];
    float dz = x[s * 3 + 2] - x[r * 3 + 2];
    float d = sqrtf(dx * dx + dy * dy + dz * dz);
    float inv = 1.f / (d + 1e-5f);
    edge4[(size_t)e * 4 + 0] = dx * inv;
    edge4[(size_t)e * 4 + 1] = dy * inv;
    edge4[(size_t)e * 4 + 2] = dz * inv;
    edge4[(size_t)e * 4 + 3] = d;
  }
}

// EA2 = [h_cat(128) | rbf(d)*hf(50) | d | zeros(13)] (192 cols, f16)
__global__ void smear_build_k(const half_t* __restrict__ EA1,
                              const half_t* __restrict__ HF16,
                              const float* __restrict__ edge4,
                              half_t* __restrict__ EA2) {
  int idx = blockIdx.x * blockDim.x + threadIdx.x;
  if (idx >= NEDGES * 192) return;
  int e = idx / 192, c = idx % 192;
  half_t out;
  if (c < 128) {
    out = EA1[(size_t)e * 128 + c];
  } else if (c < 128 + RBFN) {
    const float start = 0.006737946999085467f;           // exp(-5)
    const float wdt = (2.f / (float)RBFN) * (1.f - start);
    const float beta = 1.f / (wdt * wdt);
    int j = c - 128;
    float mean = start + (1.f - start) * (float)j / (float)(RBFN - 1);
    float d = edge4[(size_t)e * 4 + 3];
    float tt = __expf(-d) - mean;
    float rbf = __expf(-beta * tt * tt);
    out = (half_t)(rbf * (float)HF16[(size_t)e * HIDD + j]);
  } else if (c == 178) {
    out = (half_t)edge4[(size_t)e * 4 + 3];
  } else {
    out = (half_t)0.f;
  }
  EA2[(size_t)e * 192 + c] = out;
}

// sem logits (celu) + atomic max for sem & euc + degree
__global__ void attn_p1_k(const half_t* __restrict__ he16, const float* __restrict__ edge4,
                          const int* __restrict__ rcv, const float* __restrict__ Wsem,
                          const float* __restrict__ bsem, const float* __restrict__ lg,
                          float* __restrict__ sem_l, float* __restrict__ sem_max,
                          float* __restrict__ euc_max, float* __restrict__ deg) {
  int e = blockIdx.x * blockDim.x + threadIdx.x;
  if (e >= NEDGES) return;
  float acc[HEADS] = {bsem[0], bsem[1], bsem[2], bsem[3]};
  for (int j = 0; j < HIDD; ++j) {
    float hv = (float)he16[(size_t)e * HIDD + j];
    acc[0] += hv * Wsem[j * HEADS + 0];
    acc[1] += hv * Wsem[j * HEADS + 1];
    acc[2] += hv * Wsem[j * HEADS + 2];
    acc[3] += hv * Wsem[j * HEADS + 3];
  }
  int r = rcv[e];
  float d = edge4[(size_t)e * 4 + 3];
#pragma unroll
  for (int h = 0; h < HEADS; ++h) {
    float xx = acc[h];
    float c = fmaxf(xx, 0.f) + fminf(0.f, 2.f * (__expf(xx * 0.5f) - 1.f));  // celu a=2
    sem_l[(size_t)e * HEADS + h] = c;
    atomicMaxF(&sem_max[r * HEADS + h], c);
    atomicMaxF(&euc_max[r * HEADS + h], -d * __expf(lg[h]));
  }
  atomicAdd(&deg[r], 1.f);
}

__global__ void attn_p2_k(const float* __restrict__ sem_l, const float* __restrict__ edge4,
                          const int* __restrict__ rcv, const float* __restrict__ lg,
                          const float* __restrict__ sem_max, const float* __restrict__ euc_max,
                          float* __restrict__ e_sem, float* __restrict__ e_euc,
                          float* __restrict__ sem_sum, float* __restrict__ euc_sum) {
  int idx = blockIdx.x * blockDim.x + threadIdx.x;
  if (idx >= NEDGES * HEADS) return;
  int e = idx >> 2, h = idx & 3;
  int r = rcv[e];
  float ms = fixfin(sem_max[r * HEADS + h]);
  float es = __expf(sem_l[idx] - ms);
  e_sem[idx] = es;
  atomicAdd(&sem_sum[r * HEADS + h], es);
  float d = edge4[(size_t)e * 4 + 3];
  float me = fixfin(euc_max[r * HEADS + h]);
  float ee = __expf(-d * __expf(lg[h]) - me);
  e_euc[idx] = ee;
  atomicAdd(&euc_sum[r * HEADS + h], ee);
}

__global__ void attn_p3_k(const int* __restrict__ rcv, const float* __restrict__ e_sem,
                          const float* __restrict__ e_euc, const float* __restrict__ sem_sum,
                          const float* __restrict__ euc_sum, float* __restrict__ comb0,
                          float* __restrict__ comb_sum) {
  int idx = blockIdx.x * blockDim.x + threadIdx.x;
  if (idx >= NEDGES * HEADS) return;
  int e = idx >> 2, h = idx & 3;
  int r = rcv[e];
  float sn = e_sem[idx] / (sem_sum[r * HEADS + h] + 1e-9f);
  float en = e_euc[idx] / (euc_sum[r * HEADS + h] + 1e-9f);
  float c = sn * en;
  comb0[idx] = c;
  atomicAdd(&comb_sum[r * HEADS + h], c);
}

// he_att[e, j*4+h] = he[e,j]*comb[e,h];  h_agg[r] += he_att[e]
__global__ void attn_p4_k(const int* __restrict__ rcv, const half_t* __restrict__ he16,
                          const float* __restrict__ comb0, const float* __restrict__ comb_sum,
                          half_t* __restrict__ he_att16, float* __restrict__ h_agg) {
  int idx = blockIdx.x * blockDim.x + threadIdx.x;
  if (idx >= NEDGES * NCOEF) return;
  int e = idx >> 8, c = idx & 255;
  int j = c >> 2, h = c & 3;
  int r = rcv[e];
  float comb = comb0[(size_t)e * HEADS + h] / (comb_sum[r * HEADS + h] + 1e-9f);
  float val = (float)he16[(size_t)e * HIDD + j] * comb;
  he_att16[idx] = (half_t)val;
  atomicAdd(&h_agg[(size_t)r * NCOEF + c], val);
}

__global__ void comb_norm_k(const float* __restrict__ comb_vec, const float* __restrict__ deg,
                            half_t* __restrict__ out16) {
  int idx = blockIdx.x * blockDim.x + threadIdx.x;
  if (idx >= NNODES * NCOEF) return;
  int n = idx >> 8;
  float inv = 1.f / fmaxf(deg[n], 1.f);
  float s = 0.f;
#pragma unroll
  for (int a = 0; a < 3; ++a) {
    float v = comb_vec[(size_t)idx * 3 + a] * inv;
    s += v * v;
  }
  out16[idx] = (half_t)s;
}

__global__ void node_cat_k(const float* __restrict__ hn, const float* __restrict__ h_agg,
                           const half_t* __restrict__ hcomb16, half_t* __restrict__ out16) {
  int idx = blockIdx.x * blockDim.x + threadIdx.x;
  if (idx >= NNODES * 384) return;
  int n = idx / 384, c = idx % 384;
  float v;
  if (c < HIDD)            v = hn[(size_t)n * HIDD + c];
  else if (c < HIDD + NCOEF) v = h_agg[(size_t)n * NCOEF + (c - HIDD)];
  else                     v = (float)hcomb16[(size_t)n * HIDD + (c - HIDD - NCOEF)];
  out16[idx] = (half_t)v;
}

__global__ void prefix_k(const int* __restrict__ n_node, int* __restrict__ offs) {
  if (threadIdx.x == 0 && blockIdx.x == 0) {
    int s = 0;
    for (int g = 0; g < NGRAPH; ++g) { offs[g] = s; s += n_node[g]; }
    offs[NGRAPH] = s;
  }
}

__global__ void graph_sum_k(const float* __restrict__ hf, const int* __restrict__ offs,
                            float* __restrict__ y) {
  int t = blockIdx.x * blockDim.x + threadIdx.x;
  if (t >= NGRAPH * HIDD) return;
  int g = t >> 6, c = t & 63;
  int a = offs[g], b = offs[g + 1];
  if (b > NNODES) b = NNODES;
  float s = 0.f;
  for (int n = a; n < b; ++n) s += hf[(size_t)n * HIDD + c];
  y[t] = s;
}

__global__ void head0_k(const float* __restrict__ y, const float* __restrict__ W,
                        const float* __restrict__ b, float* __restrict__ th) {
  int t = blockIdx.x * blockDim.x + threadIdx.x;
  if (t >= NGRAPH * HIDD) return;
  int g = t >> 6, c = t & 63;
  float s = b[c];
  for (int j = 0; j < HIDD; ++j) s += y[g * HIDD + j] * W[j * HIDD + c];
  th[t] = siluf(s);
}

__global__ void head1_k(const float* __restrict__ th, const float* __restrict__ W,
                        const float* __restrict__ b, float* __restrict__ out) {
  int g = blockIdx.x * blockDim.x + threadIdx.x;
  if (g >= NGRAPH) return;
  float s = b[0];
  for (int j = 0; j < HIDD; ++j) s += th[g * HIDD + j] * W[j];
  out[g] = s;
}

// ---------------------------------------------------------------------------
// Host launcher
// ---------------------------------------------------------------------------
static inline int cdiv(long long a, long long b) { return (int)((a + b - 1) / b); }

extern "C" void kernel_launch(void* const* d_in, const int* in_sizes, int n_in,
                              void* d_out, int out_size, void* d_ws, size_t ws_size,
                              hipStream_t stream) {
  (void)in_sizes; (void)n_in; (void)out_size; (void)ws_size;
  const float* h        = (const float*)d_in[0];
  const float* x        = (const float*)d_in[1];
  const int*   senders  = (const int*)d_in[2];
  const int*   receivers= (const int*)d_in[3];
  const int*   n_node   = (const int*)d_in[4];
  auto P = [&](int i) -> const float* { return (const float*)d_in[i]; };
  // params flattened (jax sorted-key pytree order): embed_in{b,w}, embed_out
  // [ {b,w}, {b,w} ], head[ {b,w}, {b,w} ], layers[ 18 leaves each ]
  const float* emb_b  = P(5);  const float* emb_w  = P(6);
  const float* eo0_b  = P(7);  const float* eo0_w  = P(8);
  const float* eo1_b  = P(9);  const float* eo1_w  = P(10);
  const float* hd0_b  = P(11); const float* hd0_w  = P(12);
  const float* hd1_b  = P(13); const float* hd1_w  = P(14);
  const int LB = 15;  // per-layer leaves: ein.b, ein.w, emo0.b, emo0.w,
  // emo1.b, emo1.w, log_gamma, nm0.b, nm0.w, nm1.b, nm1.w, pn0.b, pn0.w,
  // pn1.b, pn1.w, sem.b, sem.w, xmix.w

  // ---- workspace carve ----
  char* base = (char*)d_ws;
  size_t off = 0;
  auto A = [&](size_t bytes) -> void* {
    off = (off + 255) & ~(size_t)255;
    void* p = base + off;
    off += bytes;
    return p;
  };
  float*  hn        = (float*) A((size_t)NNODES * HIDD * 4);
  half_t* hn16      = (half_t*)A((size_t)NNODES * HIDD * 2);
  half_t* EA1       = (half_t*)A((size_t)NEDGES * 128 * 2);
  float*  edge4     = (float*) A((size_t)NEDGES * 4 * 4);
  half_t* HF16      = (half_t*)A((size_t)NEDGES * HIDD * 2);
  half_t* EA2       = (half_t*)A((size_t)NEDGES * 192 * 2);
  half_t* he1_16    = (half_t*)A((size_t)NEDGES * HIDD * 2);
  half_t* he16      = (half_t*)A((size_t)NEDGES * HIDD * 2);
  float*  sem_l     = (float*) A((size_t)NEDGES * HEADS * 4);
  float*  e_sem     = (float*) A((size_t)NEDGES * HEADS * 4);
  float*  e_euc     = (float*) A((size_t)NEDGES * HEADS * 4);
  float*  comb0     = (float*) A((size_t)NEDGES * HEADS * 4);
  half_t* he_att16  = (half_t*)A((size_t)NEDGES * NCOEF * 2);
  float*  negreg    = (float*) A((size_t)2 * NNODES * HEADS * 4);   // sem_max|euc_max
  float*  zeroreg   = (float*) A((size_t)NNODES * 1037 * 4);
  half_t* cnorm16   = (half_t*)A((size_t)NNODES * NCOEF * 2);
  half_t* pn0_16    = (half_t*)A((size_t)NNODES * HIDD * 2);
  half_t* hcomb16   = (half_t*)A((size_t)NNODES * HIDD * 2);
  half_t* nodeA16   = (half_t*)A((size_t)NNODES * 384 * 2);
  half_t* nm0_16    = (half_t*)A((size_t)NNODES * HIDD * 2);
  half_t* eoA_16    = (half_t*)A((size_t)NNODES * HIDD * 2);
  float*  hfinal    = (float*) A((size_t)NNODES * HIDD * 4);
  float*  ybuf      = (float*) A((size_t)NGRAPH * HIDD * 4);
  float*  thead     = (float*) A((size_t)NGRAPH * HIDD * 4);
  int*    offs      = (int*)   A((size_t)(NGRAPH + 1) * 4);
  half_t* pw_ein[DEPTH];  half_t* pw_eo0[DEPTH]; half_t* pw_eo1[DEPTH];
  half_t* pw_xmx[DEPTH];  half_t* pw_pn0[DEPTH]; half_t* pw_pn1[DEPTH];
  half_t* pw_nm0[DEPTH];  half_t* pw_nm1[DEPTH]; float*  pb_ein[DEPTH];
  for (int l = 0; l < DEPTH; ++l) {
    pw_ein[l] = (half_t*)A(128 * 64 * 2);
    pw_eo0[l] = (half_t*)A(192 * 64 * 2);
    pw_eo1[l] = (half_t*)A(64 * 64 * 2);
    pw_xmx[l] = (half_t*)A(256 * 256 * 2);
    pw_pn0[l] = (half_t*)A(256 * 64 * 2);
    pw_pn1[l] = (half_t*)A(64 * 64 * 2);
    pw_nm0[l] = (half_t*)A(384 * 64 * 2);
    pw_nm1[l] = (half_t*)A(64 * 64 * 2);
    pb_ein[l] = (float*)A(64 * 4);
  }
  half_t* pw_embo0 = (half_t*)A(64 * 64 * 2);
  half_t* pw_embo1 = (half_t*)A(64 * 64 * 2);

  float* sem_max  = negreg;
  float* euc_max  = negreg + (size_t)NNODES * HEADS;
  float* sem_sum  = zeroreg;
  float* euc_sum  = sem_sum + (size_t)NNODES * HEADS;
  float* comb_sum = euc_sum + (size_t)NNODES * HEADS;
  float* deg      = comb_sum + (size_t)NNODES * HEADS;
  float* h_agg    = deg + NNODES;
  float* comb_vec = h_agg + (size_t)NNODES * NCOEF;

  auto pack = [&](const float* W, int K, int N, int KP, int NP, half_t* dst) {
    int total = (KP * NP) / 2;
    pack_wmma_b<<<cdiv(total, 256), 256, 0, stream>>>(W, K, N, KP, NP, dst);
  };

  // ---- weight prep (every call; deterministic) ----
  for (int l = 0; l < DEPTH; ++l) {
    const int b0 = LB + 18 * l;
    pack(P(b0 + 1),  128, RBFN, 128, 64,  pw_ein[l]);
    pack(P(b0 + 3),  179, 64,   192, 64,  pw_eo0[l]);
    pack(P(b0 + 5),  64,  64,   64,  64,  pw_eo1[l]);
    pack(P(b0 + 17), 256, 256,  256, 256, pw_xmx[l]);
    pack(P(b0 + 12), 256, 64,   256, 64,  pw_pn0[l]);
    pack(P(b0 + 14), 64,  64,   64,  64,  pw_pn1[l]);
    pack(P(b0 + 8),  384, 64,   384, 64,  pw_nm0[l]);
    pack(P(b0 + 10), 64,  64,   64,  64,  pw_nm1[l]);
    pack_bias_pad<<<1, 64, 0, stream>>>(P(b0 + 0), RBFN, 64, pb_ein[l]);
  }
  pack(eo0_w, 64, 64, 64, 64, pw_embo0);
  pack(eo1_w, 64, 64, 64, 64, pw_embo1);

  // ---- embed_in ----
  embed_in_k<<<cdiv((long long)NNODES * HIDD, 256), 256, 0, stream>>>(h, emb_w, emb_b, hn);

  const int gE32 = cdiv(NEDGES, 32);
  const int gN32 = cdiv(NNODES, 32);

  for (int l = 0; l < DEPTH; ++l) {
    const int b0 = LB + 18 * l;
    const float* lg     = P(b0 + 6);
    const float* emo0_b = P(b0 + 2);
    const float* emo1_b = P(b0 + 4);
    const float* nm0_b  = P(b0 + 7);
    const float* nm1_b  = P(b0 + 9);
    const float* pn0_b  = P(b0 + 11);
    const float* pn1_b  = P(b0 + 13);
    const float* sem_b  = P(b0 + 15);
    const float* sem_w  = P(b0 + 16);

    fill_f32<<<cdiv((long long)2 * NNODES * HEADS, 256), 256, 0, stream>>>(
        negreg, -__builtin_huge_valf(), 2 * NNODES * HEADS);
    fill_f32<<<cdiv((long long)NNODES * 1037, 256), 256, 0, stream>>>(
        zeroreg, 0.f, NNODES * 1037);

    edge_build_k<<<cdiv((long long)NEDGES * 16, 256), 256, 0, stream>>>(
        hn, x, senders, receivers, EA1, edge4);
    gemm_rows<128, 64><<<gE32, 256, 0, stream>>>(
        EA1, pw_ein[l], pb_ein[l], NEDGES, nullptr, HF16, nullptr, 0);
    smear_build_k<<<cdiv((long long)NEDGES * 192, 256), 256, 0, stream>>>(
        EA1, HF16, edge4, EA2);
    gemm_rows<192, 64><<<gE32, 256, 0, stream>>>(
        EA2, pw_eo0[l], emo0_b, NEDGES, nullptr, he1_16, nullptr, 1);
    gemm_rows<64, 64><<<gE32, 256, 0, stream>>>(
        he1_16, pw_eo1[l], emo1_b, NEDGES, nullptr, he16, nullptr, 1);

    attn_p1_k<<<cdiv(NEDGES, 256), 256, 0, stream>>>(
        he16, edge4, receivers, sem_w, sem_b, lg, sem_l, sem_max, euc_max, deg);
    attn_p2_k<<<cdiv((long long)NEDGES * HEADS, 256), 256, 0, stream>>>(
        sem_l, edge4, receivers, lg, sem_max, euc_max, e_sem, e_euc, sem_sum, euc_sum);
    attn_p3_k<<<cdiv((long long)NEDGES * HEADS, 256), 256, 0, stream>>>(
        receivers, e_sem, e_euc, sem_sum, euc_sum, comb0, comb_sum);
    attn_p4_k<<<cdiv((long long)NEDGES * NCOEF, 256), 256, 0, stream>>>(
        receivers, he16, comb0, comb_sum, he_att16, h_agg);

    xmix_scatter<<<gE32, 256, 0, stream>>>(
        he_att16, pw_xmx[l], receivers, edge4, comb_vec, NEDGES);

    comb_norm_k<<<cdiv((long long)NNODES * NCOEF, 256), 256, 0, stream>>>(
        comb_vec, deg, cnorm16);
    gemm_rows<256, 64><<<gN32, 256, 0, stream>>>(
        cnorm16, pw_pn0[l], pn0_b, NNODES, nullptr, pn0_16, nullptr, 1);
    gemm_rows<64, 64><<<gN32, 256, 0, stream>>>(
        pn0_16, pw_pn1[l], pn1_b, NNODES, nullptr, hcomb16, nullptr, 1);
    node_cat_k<<<cdiv((long long)NNODES * 384, 256), 256, 0, stream>>>(
        hn, h_agg, hcomb16, nodeA16);
    gemm_rows<384, 64><<<gN32, 256, 0, stream>>>(
        nodeA16, pw_nm0[l], nm0_b, NNODES, nullptr, nm0_16, nullptr, 1);
    gemm_rows<64, 64><<<gN32, 256, 0, stream>>>(
        nm0_16, pw_nm1[l], nm1_b, NNODES, hn, hn16, hn, 1);  // hn += silu(...)
  }

  // ---- readout ----
  gemm_rows<64, 64><<<gN32, 256, 0, stream>>>(
      hn16, pw_embo0, eo0_b, NNODES, nullptr, eoA_16, nullptr, 1);
  gemm_rows<64, 64><<<gN32, 256, 0, stream>>>(
      eoA_16, pw_embo1, eo1_b, NNODES, hfinal, nullptr, nullptr, 0);
  prefix_k<<<1, 32, 0, stream>>>(n_node, offs);
  graph_sum_k<<<cdiv((long long)NGRAPH * HIDD, 256), 256, 0, stream>>>(hfinal, offs, ybuf);
  head0_k<<<cdiv((long long)NGRAPH * HIDD, 256), 256, 0, stream>>>(ybuf, hd0_w, hd0_b, thead);
  head1_k<<<1, 128, 0, stream>>>(thead, hd1_w, hd1_b, (float*)d_out);
}
